// SelfAttentionLayer_42107859370724
// MI455X (gfx1250) — matrix operand order repaired and checked by reading, not compiled, not run on hardware
//
#include <hip/hip_runtime.h>

// ------------------------------ types ---------------------------------
typedef __bf16 bf16;
typedef __bf16 bf16x16 __attribute__((ext_vector_type(16)));
typedef float  f32x8   __attribute__((ext_vector_type(8)));
typedef float  f32x4   __attribute__((ext_vector_type(4)));
typedef int    i32x4   __attribute__((ext_vector_type(4)));

union FragBF {
  bf16x16 v;
  i32x4   q[2];
};

__device__ __forceinline__ bf16 f2bf(float x) {
  unsigned u = __float_as_uint(x);
  u += 0x7fffu + ((u >> 16) & 1u);          // round-to-nearest-even
  unsigned short s = (unsigned short)(u >> 16);
  return __builtin_bit_cast(bf16, s);
}

#define WMMA_BF16(A, B, C) \
  __builtin_amdgcn_wmma_f32_16x16x32_bf16(false, (A), false, (B), (short)0, (C), false, false)

// CDNA5 async global->LDS copy (16B per lane), tracked by ASYNCcnt.
__device__ __forceinline__ void async_copy_b128(void* lds_dst, const void* gsrc) {
  unsigned l = (unsigned)(size_t)lds_dst;                 // LDS byte address
  unsigned long long g = (unsigned long long)(size_t)gsrc;
  asm volatile("global_load_async_to_lds_b128 %0, %1, off"
               :: "v"(l), "v"(g) : "memory");
}
__device__ __forceinline__ void wait_asynccnt0() {
  asm volatile("s_wait_asynccnt 0x0" ::: "memory");
}

#define DMODEL 1024
#define SEQ    2048
#define NHEAD  16
#define HDIM   64
#define NROWS  4096   // B * SEQ

// ---------------------------------------------------------------------
// Kernel 1: fused QKV projection.  C[r,o] = X[r,:] . W[o,:] + b[o]
// 64x64 tile per block, 8 waves, each wave: one 16x16 A-subtile x two
// N-subtiles.  fp32 -> bf16 conversion during LDS staging.
// ---------------------------------------------------------------------
__global__ __launch_bounds__(256)
void qkv_gemm_kernel(const float* __restrict__ X,
                     const float* __restrict__ Wq, const float* __restrict__ Bq,
                     const float* __restrict__ Wk, const float* __restrict__ Bk,
                     const float* __restrict__ Wv, const float* __restrict__ Bv,
                     bf16* __restrict__ Qo, bf16* __restrict__ Ko, bf16* __restrict__ Vo,
                     float qscale)
{
  __shared__ __align__(16) bf16 lds_a[64][40];
  __shared__ __align__(16) bf16 lds_b[64][40];

  const int t    = threadIdx.x;
  const int wave = t >> 5;
  const int lane = t & 31;
  const int nk   = lane & 15;
  const int kg8  = (lane >> 4) * 8;
  const int h16  = (lane >> 4) * 16;

  const float* W; const float* Bi; bf16* Out; float scale;
  if (blockIdx.z == 0)      { W = Wq; Bi = Bq; Out = Qo; scale = qscale; }
  else if (blockIdx.z == 1) { W = Wk; Bi = Bk; Out = Ko; scale = 1.0f; }
  else                      { W = Wv; Bi = Bv; Out = Vo; scale = 1.0f; }

  const int m0 = blockIdx.y * 64;   // token rows
  const int n0 = blockIdx.x * 64;   // output features
  const int mi = wave & 3;          // M subtile
  const int nj = (wave >> 2) * 2;   // first of two N subtiles

  f32x8 acc0 = {}; f32x8 acc1 = {};

  for (int k0 = 0; k0 < DMODEL; k0 += 32) {
    #pragma unroll
    for (int it = 0; it < 2; ++it) {
      int idx = (t + it * 256) * 4;
      int r = idx >> 5;
      int c = idx & 31;
      f32x4 xa = *(const f32x4*)(X + (size_t)(m0 + r) * DMODEL + k0 + c);
      f32x4 xb = *(const f32x4*)(W + (size_t)(n0 + r) * DMODEL + k0 + c);
      #pragma unroll
      for (int e = 0; e < 4; ++e) {
        lds_a[r][c + e] = f2bf(xa[e]);
        lds_b[r][c + e] = f2bf(xb[e]);
      }
      if (k0 + 32 < DMODEL) {   // warm L2->WGP path for next K-step
        __builtin_prefetch(X + (size_t)(m0 + r) * DMODEL + k0 + 32 + c, 0, 1);
        __builtin_prefetch(W + (size_t)(n0 + r) * DMODEL + k0 + 32 + c, 0, 1);
      }
    }
    __syncthreads();

    FragBF a, b0, b1;
    const int arow = mi * 16 + nk;
    a.q[0] = *(const i32x4*)&lds_a[arow][kg8];
    a.q[1] = *(const i32x4*)&lds_a[arow][16 + kg8];
    const int c0 = (nj + 0) * 16 + nk;
    const int c1 = (nj + 1) * 16 + nk;
    b0.q[0] = *(const i32x4*)&lds_b[c0][kg8];
    b0.q[1] = *(const i32x4*)&lds_b[c0][16 + kg8];
    b1.q[0] = *(const i32x4*)&lds_b[c1][kg8];
    b1.q[1] = *(const i32x4*)&lds_b[c1][16 + kg8];

    acc0 = WMMA_BF16(a.v, b0.v, acc0);
    acc1 = WMMA_BF16(a.v, b1.v, acc1);
    __syncthreads();
  }

  // epilogue: bias (+scale) then scatter to [B*H, S, HD] bf16
  const int rowbase = m0 + mi * 16 + (lane >> 4) * 8;
  #pragma unroll
  for (int v = 0; v < 8; ++v) {
    int r  = rowbase + v;
    int bb = r >> 11;           // / SEQ
    int s  = r & 2047;
    int o0 = n0 + (nj + 0) * 16 + nk;
    int o1 = n0 + (nj + 1) * 16 + nk;
    float v0 = (acc0[v] + Bi[o0]) * scale;
    float v1 = (acc1[v] + Bi[o1]) * scale;
    Out[((size_t)(bb * NHEAD + (o0 >> 6)) * SEQ + s) * HDIM + (o0 & 63)] = f2bf(v0);
    Out[((size_t)(bb * NHEAD + (o1 >> 6)) * SEQ + s) * HDIM + (o1 & 63)] = f2bf(v1);
  }
}

// ---------------------------------------------------------------------
// Kernel 2: causal flash attention.  128 queries / block, 8 waves x 16
// rows, 32-key chunks.  K tile filled with CDNA5 async global->LDS
// copies (ASYNCcnt), scores + PV via WMMA, online softmax via
// __shfl_xor reductions over the 16-lane C-matrix groups.
// ---------------------------------------------------------------------
__global__ __launch_bounds__(256)
void attention_kernel(const bf16* __restrict__ Q, const bf16* __restrict__ K,
                      const bf16* __restrict__ V, bf16* __restrict__ Ctx)
{
  __shared__ __align__(16) bf16 k_lds[32][72];        // [key][d]
  __shared__ __align__(16) bf16 vt_lds[64][40];       // [d][key]
  __shared__ __align__(16) bf16 p_lds[8][16][40];     // per-wave P tile

  const int t    = threadIdx.x;
  const int wave = t >> 5;
  const int lane = t & 31;
  const int nk   = lane & 15;
  const int kg8  = (lane >> 4) * 8;
  const int h16  = (lane >> 4) * 16;

  const int bh    = blockIdx.y;    // b*16 + h
  const int bb    = bh >> 4;
  const int hh    = bh & 15;
  const int q0blk = blockIdx.x * 128;
  const int qbase = q0blk + wave * 16;

  const bf16* Qh = Q + (size_t)bh * SEQ * HDIM;
  const bf16* Kh = K + (size_t)bh * SEQ * HDIM;
  const bf16* Vh = V + (size_t)bh * SEQ * HDIM;

  // Q A-fragments for d-chunks {0..31} and {32..63}
  FragBF qa0, qa1;
  {
    const bf16* qrow = Qh + (size_t)(qbase + nk) * HDIM;
    qa0.q[0] = *(const i32x4*)(qrow + kg8);
    qa0.q[1] = *(const i32x4*)(qrow + 16 + kg8);
    qa1.q[0] = *(const i32x4*)(qrow + 32 + kg8);
    qa1.q[1] = *(const i32x4*)(qrow + 48 + kg8);
  }

  f32x8 o0 = {}, o1 = {}, o2 = {}, o3 = {};
  float mrow[8], lrow[8];
  #pragma unroll
  for (int v = 0; v < 8; ++v) { mrow[v] = -1e30f; lrow[v] = 0.0f; }

  const int nchunks = ((q0blk + 127) >> 5) + 1;    // causal limit of block
  for (int c = 0; c < nchunks; ++c) {
    const int kc = c * 32;
    {   // K chunk (32x64 bf16): async global->LDS, one b128 per thread
      int idx = t * 8;
      int key = idx >> 6;
      int d   = idx & 63;
      async_copy_b128(&k_lds[key][d], Kh + (size_t)(kc + key) * HDIM + d);
      if (c + 1 < nchunks) {   // prefetch next chunk's K/V
        __builtin_prefetch(Kh + (size_t)(kc + 32 + key) * HDIM + d, 0, 1);
        __builtin_prefetch(Vh + (size_t)(kc + 32 + key) * HDIM + d, 0, 1);
      }
    }
    #pragma unroll
    for (int i = 0; i < 8; ++i) {   // stage V transposed (VGPR path)
      int e   = t + i * 256;
      int key = e >> 6;
      int d   = e & 63;
      vt_lds[d][key] = Vh[(size_t)(kc + key) * HDIM + d];
    }
    wait_asynccnt0();
    __syncthreads();

    if (kc <= qbase + 15) {   // chunk visible to this wave's queries
      // ---- scores: S = Q . K^T  (two 16-key halves, two d-chunks) ----
      FragBF kb00, kb01, kb10, kb11;
      kb00.q[0] = *(const i32x4*)&k_lds[nk][kg8];
      kb00.q[1] = *(const i32x4*)&k_lds[nk][16 + kg8];
      kb01.q[0] = *(const i32x4*)&k_lds[nk][32 + kg8];
      kb01.q[1] = *(const i32x4*)&k_lds[nk][48 + kg8];
      kb10.q[0] = *(const i32x4*)&k_lds[16 + nk][kg8];
      kb10.q[1] = *(const i32x4*)&k_lds[16 + nk][16 + kg8];
      kb11.q[0] = *(const i32x4*)&k_lds[16 + nk][32 + kg8];
      kb11.q[1] = *(const i32x4*)&k_lds[16 + nk][48 + kg8];

      f32x8 s0 = {}, s1 = {};
      s0 = WMMA_BF16(qa0.v, kb00.v, s0);
      s0 = WMMA_BF16(qa1.v, kb01.v, s0);
      s1 = WMMA_BF16(qa0.v, kb10.v, s1);
      s1 = WMMA_BF16(qa1.v, kb11.v, s1);

      // ---- causal mask + online softmax ----
      const int qr0 = qbase + (lane >> 4) * 8;
      float fsc[8];
      #pragma unroll
      for (int v = 0; v < 8; ++v) {
        int qi = qr0 + v;
        if (kc + nk > qi)      s0[v] = -1e30f;
        if (kc + 16 + nk > qi) s1[v] = -1e30f;
        float x = fmaxf(s0[v], s1[v]);
        x = fmaxf(x, __shfl_xor(x, 1, 32));
        x = fmaxf(x, __shfl_xor(x, 2, 32));
        x = fmaxf(x, __shfl_xor(x, 4, 32));
        x = fmaxf(x, __shfl_xor(x, 8, 32));
        float nm = fmaxf(mrow[v], x);
        fsc[v]  = __expf(mrow[v] - nm);
        mrow[v] = nm;
        float p0 = __expf(s0[v] - nm);
        float p1 = __expf(s1[v] - nm);
        float rs = p0 + p1;
        rs += __shfl_xor(rs, 1, 32);
        rs += __shfl_xor(rs, 2, 32);
        rs += __shfl_xor(rs, 4, 32);
        rs += __shfl_xor(rs, 8, 32);
        lrow[v] = lrow[v] * fsc[v] + rs;
        int M = v + (lane >> 4) * 8;          // C-layout row for this lane
        p_lds[wave][M][nk]      = f2bf(p0);
        p_lds[wave][M][16 + nk] = f2bf(p1);
        o0[v] *= fsc[v]; o1[v] *= fsc[v]; o2[v] *= fsc[v]; o3[v] *= fsc[v];
      }

      // P back out of LDS in A-fragment layout (K = 32 keys)
      FragBF pa;
      pa.q[0] = *(const i32x4*)&p_lds[wave][nk][kg8];
      pa.q[1] = *(const i32x4*)&p_lds[wave][nk][16 + kg8];

      // ---- O += P . V  (four 16-wide d subtiles) ----
      FragBF vb;
      vb.q[0] = *(const i32x4*)&vt_lds[0 * 16 + nk][kg8];
      vb.q[1] = *(const i32x4*)&vt_lds[0 * 16 + nk][16 + kg8];
      o0 = WMMA_BF16(pa.v, vb.v, o0);
      vb.q[0] = *(const i32x4*)&vt_lds[1 * 16 + nk][kg8];
      vb.q[1] = *(const i32x4*)&vt_lds[1 * 16 + nk][16 + kg8];
      o1 = WMMA_BF16(pa.v, vb.v, o1);
      vb.q[0] = *(const i32x4*)&vt_lds[2 * 16 + nk][kg8];
      vb.q[1] = *(const i32x4*)&vt_lds[2 * 16 + nk][16 + kg8];
      o2 = WMMA_BF16(pa.v, vb.v, o2);
      vb.q[0] = *(const i32x4*)&vt_lds[3 * 16 + nk][kg8];
      vb.q[1] = *(const i32x4*)&vt_lds[3 * 16 + nk][16 + kg8];
      o3 = WMMA_BF16(pa.v, vb.v, o3);
    }
    __syncthreads();
  }

  // normalize, write context back in [B, S, D] bf16 layout
  #pragma unroll
  for (int v = 0; v < 8; ++v) {
    float inv = 1.0f / lrow[v];
    int qi = qbase + v + (lane >> 4) * 8;
    bf16* dst = Ctx + (size_t)(bb * SEQ + qi) * DMODEL + hh * HDIM;
    dst[0 * 16 + nk] = f2bf(o0[v] * inv);
    dst[1 * 16 + nk] = f2bf(o1[v] * inv);
    dst[2 * 16 + nk] = f2bf(o2[v] * inv);
    dst[3 * 16 + nk] = f2bf(o3[v] * inv);
  }
}

// ---------------------------------------------------------------------
// Kernel 3: output projection + bias + residual (f32 out to workspace)
// ---------------------------------------------------------------------
__global__ __launch_bounds__(256)
void out_gemm_kernel(const bf16* __restrict__ A, const float* __restrict__ Wo,
                     const float* __restrict__ Bo, const float* __restrict__ Resid,
                     float* __restrict__ Y)
{
  __shared__ __align__(16) bf16 lds_a[64][40];
  __shared__ __align__(16) bf16 lds_b[64][40];

  const int t    = threadIdx.x;
  const int wave = t >> 5;
  const int lane = t & 31;
  const int nk   = lane & 15;
  const int kg8  = (lane >> 4) * 8;
  const int h16  = (lane >> 4) * 16;

  const int m0 = blockIdx.y * 64;
  const int n0 = blockIdx.x * 64;
  const int mi = wave & 3;
  const int nj = (wave >> 2) * 2;

  f32x8 acc0 = {}, acc1 = {};

  for (int k0 = 0; k0 < DMODEL; k0 += 32) {
    {   // A already bf16: async global->LDS b128 copy per thread
      int idx = t * 8;
      int r = idx >> 5;
      int cc = idx & 31;
      async_copy_b128(&lds_a[r][cc], A + (size_t)(m0 + r) * DMODEL + k0 + cc);
    }
    #pragma unroll
    for (int it = 0; it < 2; ++it) {
      int idx = (t + it * 256) * 4;
      int r = idx >> 5;
      int cc = idx & 31;
      f32x4 xb = *(const f32x4*)(Wo + (size_t)(n0 + r) * DMODEL + k0 + cc);
      #pragma unroll
      for (int e = 0; e < 4; ++e) lds_b[r][cc + e] = f2bf(xb[e]);
      if (k0 + 32 < DMODEL)
        __builtin_prefetch(Wo + (size_t)(n0 + r) * DMODEL + k0 + 32 + cc, 0, 1);
    }
    wait_asynccnt0();
    __syncthreads();

    FragBF a, b0, b1;
    const int arow = mi * 16 + nk;
    a.q[0] = *(const i32x4*)&lds_a[arow][kg8];
    a.q[1] = *(const i32x4*)&lds_a[arow][16 + kg8];
    const int c0 = (nj + 0) * 16 + nk;
    const int c1 = (nj + 1) * 16 + nk;
    b0.q[0] = *(const i32x4*)&lds_b[c0][kg8];
    b0.q[1] = *(const i32x4*)&lds_b[c0][16 + kg8];
    b1.q[0] = *(const i32x4*)&lds_b[c1][kg8];
    b1.q[1] = *(const i32x4*)&lds_b[c1][16 + kg8];

    acc0 = WMMA_BF16(a.v, b0.v, acc0);
    acc1 = WMMA_BF16(a.v, b1.v, acc1);
    __syncthreads();
  }

  const int rowbase = m0 + mi * 16 + (lane >> 4) * 8;
  #pragma unroll
  for (int v = 0; v < 8; ++v) {
    int r  = rowbase + v;
    int o0 = n0 + (nj + 0) * 16 + nk;
    int o1 = n0 + (nj + 1) * 16 + nk;
    Y[(size_t)r * DMODEL + o0] = acc0[v] + Bo[o0] + Resid[(size_t)r * DMODEL + o0];
    Y[(size_t)r * DMODEL + o1] = acc1[v] + Bo[o1] + Resid[(size_t)r * DMODEL + o1];
  }
}

// ---------------------------------------------------------------------
// Kernel 4: LayerNorm over the last dim (1024), one block per row
// ---------------------------------------------------------------------
__global__ __launch_bounds__(256)
void ln_kernel(const float* __restrict__ Y, const float* __restrict__ G,
               const float* __restrict__ Bt, float* __restrict__ Out)
{
  __shared__ float ps[8], pss[8];
  const int row = blockIdx.x;
  const float* y = Y + (size_t)row * DMODEL;

  float vals[4];
  float s = 0.0f, ss = 0.0f;
  #pragma unroll
  for (int i = 0; i < 4; ++i) {
    float x = y[threadIdx.x + i * 256];
    vals[i] = x; s += x; ss += x * x;
  }
  #pragma unroll
  for (int m = 1; m < 32; m <<= 1) {
    s  += __shfl_xor(s,  m, 32);
    ss += __shfl_xor(ss, m, 32);
  }
  if ((threadIdx.x & 31) == 0) { ps[threadIdx.x >> 5] = s; pss[threadIdx.x >> 5] = ss; }
  __syncthreads();
  float ts = 0.0f, tss = 0.0f;
  #pragma unroll
  for (int w = 0; w < 8; ++w) { ts += ps[w]; tss += pss[w]; }
  const float mu  = ts * (1.0f / DMODEL);
  const float var = tss * (1.0f / DMODEL) - mu * mu;
  const float r   = rsqrtf(var + 1e-5f);
  #pragma unroll
  for (int i = 0; i < 4; ++i) {
    int o = threadIdx.x + i * 256;
    Out[(size_t)row * DMODEL + o] = (vals[i] - mu) * r * G[o] + Bt[o];
  }
}

// ---------------------------------------------------------------------
extern "C" void kernel_launch(void* const* d_in, const int* in_sizes, int n_in,
                              void* d_out, int out_size, void* d_ws, size_t ws_size,
                              hipStream_t stream)
{
  (void)in_sizes; (void)n_in; (void)out_size; (void)ws_size;

  const float* hidden = (const float*)d_in[0];
  const float* Wq = (const float*)d_in[1];
  const float* bq = (const float*)d_in[2];
  const float* Wk = (const float*)d_in[3];
  const float* bk = (const float*)d_in[4];
  const float* Wv = (const float*)d_in[5];
  const float* bv = (const float*)d_in[6];
  const float* Wo = (const float*)d_in[7];
  const float* bo = (const float*)d_in[8];
  const float* ln_g = (const float*)d_in[9];
  const float* ln_b = (const float*)d_in[10];
  float* out = (float*)d_out;

  char* ws = (char*)d_ws;
  size_t off = 0;
  bf16* q16   = (bf16*)(ws + off); off += (size_t)NROWS * DMODEL * sizeof(bf16);
  bf16* k16   = (bf16*)(ws + off); off += (size_t)NROWS * DMODEL * sizeof(bf16);
  bf16* v16   = (bf16*)(ws + off); off += (size_t)NROWS * DMODEL * sizeof(bf16);
  bf16* ctx16 = (bf16*)(ws + off); off += (size_t)NROWS * DMODEL * sizeof(bf16);
  float* y32  = (float*)(ws + off); off += (size_t)NROWS * DMODEL * sizeof(float);

  const float qscale = 0.125f;   // HD^-0.5 = 64^-0.5

  qkv_gemm_kernel<<<dim3(16, 64, 3), 256, 0, stream>>>(
      hidden, Wq, bq, Wk, bk, Wv, bv, q16, k16, v16, qscale);

  attention_kernel<<<dim3(16, 32), 256, 0, stream>>>(q16, k16, v16, ctx16);

  out_gemm_kernel<<<dim3(16, 64), 256, 0, stream>>>(ctx16, Wo, bo, hidden, y32);

  ln_kernel<<<NROWS, 256, 0, stream>>>(y32, ln_g, ln_b, out);
}
